// MultiheadAttention_88785563943047
// MI455X (gfx1250) — matrix operand order repaired and checked by reading, not compiled, not run on hardware
//
#include <hip/hip_runtime.h>
#include <hip/hip_bf16.h>
#include <math.h>
#include <stdint.h>

// ---------------------------------------------------------------------------
// MI455X decode-attention step.
//   H=32 heads, DQ=DK=DV=128, DIN=DOUT=4096, S=16384 cache rows, seq=1.
// Bandwidth-bound: ~1.3 GB of traffic (weights 256MB read, cache copy
// 512MB read + 512MB write) at 23.3 TB/s => ~56us floor. Matvecs go through
// v_wmma_f32_16x16x32_f16 (wave32, 16 outputs/wave, 2 WMMA per loop trip);
// the cache copy uses the CDNA5 async LDS path (ASYNCcnt): global load ->
// LDS -> global store without touching VGPRs, waits via s_wait_asynccnt.
// ---------------------------------------------------------------------------

#define HEADS 32
#define DHEAD 128
#define DIN   4096
#define DOUT  4096
#define SLEN  16384
#define SEQ   1            // x is (1, DIN); reference mask tril((SEQ,S)) -> only j<SEQ valid
#define THETA 10000.0f

typedef __attribute__((ext_vector_type(16))) _Float16 v16h;
typedef __attribute__((ext_vector_type(8)))  float    v8f;
typedef __attribute__((ext_vector_type(4)))  int      v4i;

#define GAS __attribute__((address_space(1)))
#define LAS __attribute__((address_space(3)))

#if defined(__has_builtin)
#  if __has_builtin(__builtin_amdgcn_global_load_async_to_lds_b128) && \
      __has_builtin(__builtin_amdgcn_global_store_async_from_lds_b128)
#    define HAVE_ASYNC_LDS 1
#  endif
#endif
#ifndef HAVE_ASYNC_LDS
#  define HAVE_ASYNC_LDS 0
#endif

__device__ __forceinline__ void wait_async0() {
#if defined(__has_builtin) && __has_builtin(__builtin_amdgcn_s_wait_asynccnt)
    __builtin_amdgcn_s_wait_asynccnt(0);
#else
    asm volatile("s_wait_asynccnt 0x0" ::: "memory");
#endif
}

// ---------------------------------------------------------------------------
// WMMA matvec: y[row0 + n] = dot(x[0:K], W[row0+n][0:K]),  n = 0..15.
// One wave (32 threads) per 16 output rows. The x chunk (32 values) is
// broadcast into every A-matrix row in contiguous k order
// (lanes 0-15: K=0..15 ; lanes 16-31: K=16..31), so all 16 D rows
// hold the same dot products; we read D row 0 (acc[0], lanes 0..15).
// B holds 16 weight rows: lane n = column n, VGPR j packs K = half*16 + 2j.
// Each lane streams 16 consecutive floats (64B) of its weight row per step;
// the K loop is unrolled x2 (two v_wmma per trip, 128B/lane in flight).
// ---------------------------------------------------------------------------
__device__ __forceinline__ v16h build_a(const float* __restrict__ x, int k0, int half) {
    v16h a;
#pragma unroll
    for (int j = 0; j < 4; ++j) {
        const int k = k0 + half * 16 + 2 * j;
        a[2 * j]     = (_Float16)x[k];
        a[2 * j + 1] = (_Float16)x[k + 1];
    }
#pragma unroll
    for (int j = 0; j < 4; ++j) {
        const int k = k0 + half * 16 + 8 + 2 * j;
        a[8 + 2 * j]     = (_Float16)x[k];
        a[8 + 2 * j + 1] = (_Float16)x[k + 1];
    }
    return a;
}

__device__ __forceinline__ v16h build_b(const float* __restrict__ wrow, int k0, int half) {
    const float4* wr4 = reinterpret_cast<const float4*>(wrow + k0 + half * 16);
    float4 w0 = wr4[0], w1 = wr4[1], w2 = wr4[2], w3 = wr4[3];
    v16h b;
    b[0]  = (_Float16)w0.x;  b[1]  = (_Float16)w0.y;
    b[2]  = (_Float16)w0.z;  b[3]  = (_Float16)w0.w;
    b[4]  = (_Float16)w1.x;  b[5]  = (_Float16)w1.y;
    b[6]  = (_Float16)w1.z;  b[7]  = (_Float16)w1.w;
    b[8]  = (_Float16)w2.x;  b[9]  = (_Float16)w2.y;
    b[10] = (_Float16)w2.z;  b[11] = (_Float16)w2.w;
    b[12] = (_Float16)w3.x;  b[13] = (_Float16)w3.y;
    b[14] = (_Float16)w3.z;  b[15] = (_Float16)w3.w;
    return b;
}

__global__ __launch_bounds__(32)
void wmma_matvec_kernel(const float* __restrict__ x,
                        const float* __restrict__ W,
                        float* __restrict__ y, int K) {
    const int lane = threadIdx.x;            // 0..31
    const int row0 = blockIdx.x * 16;
    const int half = lane >> 4;              // 0: lanes 0-15, 1: lanes 16-31
    const int n    = lane & 15;              // output column within tile

    v8f acc = {};                            // C/D accumulator (8 VGPRs)
    const float* wrow = W + (size_t)(row0 + n) * (size_t)K;

    for (int k0 = 0; k0 < K; k0 += 64) {
        v16h a0 = build_a(x, k0, half);
        v16h b0 = build_b(wrow, k0, half);
        acc = __builtin_amdgcn_wmma_f32_16x16x32_f16(
            false, a0, false, b0, (short)0, acc, false, false);

        v16h a1 = build_a(x, k0 + 32, half);
        v16h b1 = build_b(wrow, k0 + 32, half);
        acc = __builtin_amdgcn_wmma_f32_16x16x32_f16(
            false, a1, false, b1, (short)0, acc, false, false);
    }

    // D row 0 lives in acc[0], lanes 0..15 (M=0, N=lane).
    if (lane < 16) y[row0 + lane] = acc[0];
}

// ---------------------------------------------------------------------------
// RoPE for q and k (per-head, positions 0..SEQ-1; SEQ=1 -> position 0).
// ---------------------------------------------------------------------------
__global__ __launch_bounds__(64)
void rope_kernel(float* __restrict__ q, float* __restrict__ k) {
    const int h = blockIdx.x;
    const int i = threadIdx.x;                 // 0..63 (half = DHEAD/2)
    const float inv_freq = __powf(THETA, -((float)i) / 64.0f);
    const float ang = 0.0f * inv_freq;         // position s = 0 for seq==1
    const float c = __cosf(ang);
    const float s = __sinf(ang);

    float* qh = q + h * DHEAD;
    float* kh = k + h * DHEAD;

    float q1 = qh[i], q2 = qh[i + 64];
    qh[i]      = q1 * c - q2 * s;
    qh[i + 64] = q2 * c + q1 * s;

    float k1 = kh[i], k2 = kh[i + 64];
    kh[i]      = k1 * c - k2 * s;
    kh[i + 64] = k2 * c + k1 * s;
}

// ---------------------------------------------------------------------------
// Bulk cache copy: d_out caches = input caches (row update done separately).
// Preferred path: CDNA5 async global<->LDS transfers (ASYNCcnt). The builtins
// are typed on v4i (b128) pointers: global AS(1) src/dst + LDS AS(3) slot.
// With ~16K waves resident, the per-wave load->wait->store chain is hidden.
// ---------------------------------------------------------------------------
#if HAVE_ASYNC_LDS
__global__ __launch_bounds__(256)
void copy_caches_kernel(const float4* __restrict__ kin,
                        const float4* __restrict__ vin,
                        float4* __restrict__ kout,
                        float4* __restrict__ vout) {
    __shared__ float4 stage[2][256];           // 8KB: ping (key) / pong (value)
    const size_t n4     = (size_t)SLEN * HEADS * DHEAD / 4;
    const size_t stride = (size_t)gridDim.x * blockDim.x;
    const unsigned t    = threadIdx.x;

    // Per-lane private LDS slots (generic -> 32-bit LDS offset -> AS(3) v4i*).
    LAS v4i* lk = (LAS v4i*)(uint32_t)(uintptr_t)&stage[0][t];
    LAS v4i* lv = (LAS v4i*)(uint32_t)(uintptr_t)&stage[1][t];

    for (size_t i = (size_t)blockIdx.x * blockDim.x + t; i < n4; i += stride) {
        GAS v4i* gks = (GAS v4i*)(uintptr_t)(kin + i);
        GAS v4i* gvs = (GAS v4i*)(uintptr_t)(vin + i);
        GAS v4i* gkd = (GAS v4i*)(uintptr_t)(kout + i);
        GAS v4i* gvd = (GAS v4i*)(uintptr_t)(vout + i);

        __builtin_amdgcn_global_load_async_to_lds_b128(gks, lk, 0, 0);
        __builtin_amdgcn_global_load_async_to_lds_b128(gvs, lv, 0, 0);
        wait_async0();                                   // loads landed in LDS
        __builtin_amdgcn_global_store_async_from_lds_b128(gkd, lk, 0, 0);
        __builtin_amdgcn_global_store_async_from_lds_b128(gvd, lv, 0, 0);
        wait_async0();                                   // LDS reusable
    }
}
#else
__global__ __launch_bounds__(256)
void copy_caches_kernel(const float4* __restrict__ kin,
                        const float4* __restrict__ vin,
                        float4* __restrict__ kout,
                        float4* __restrict__ vout) {
    const size_t n4     = (size_t)SLEN * HEADS * DHEAD / 4;
    const size_t stride = (size_t)gridDim.x * blockDim.x;
    for (size_t i = (size_t)blockIdx.x * blockDim.x + threadIdx.x; i < n4; i += stride) {
        __builtin_prefetch(&kin[i + 4 * stride], 0, 1);
        __builtin_prefetch(&vin[i + 4 * stride], 0, 1);
        kout[i] = kin[i];
        vout[i] = vin[i];
    }
}
#endif

// ---------------------------------------------------------------------------
// Overwrite cache row `index` with the fresh k / v vectors (4096 floats each).
// ---------------------------------------------------------------------------
__global__ __launch_bounds__(256)
void update_row_kernel(const float4* __restrict__ knew,
                       const float4* __restrict__ vnew,
                       const int* __restrict__ indexp,
                       float4* __restrict__ kout,
                       float4* __restrict__ vout) {
    const size_t rowlen4 = (size_t)HEADS * DHEAD / 4;    // 1024 float4
    const size_t base    = (size_t)(*indexp) * rowlen4;
    for (size_t c = (size_t)blockIdx.x * blockDim.x + threadIdx.x;
         c < rowlen4; c += (size_t)gridDim.x * blockDim.x) {
        kout[base + c] = knew[c];
        vout[base + c] = vnew[c];
    }
}

// ---------------------------------------------------------------------------
// Masked attention. Reference: mask = tril(ones((SEQ,S))), so for query row 0
// only positions j < SEQ are live; everything else is -inf before softmax.
// Compute logits over the valid prefix, exact softmax, accumulate V.
// ---------------------------------------------------------------------------
__global__ __launch_bounds__(128)
void attn_kernel(const float* __restrict__ q,     // (H, DHEAD), un-scaled
                 const float* __restrict__ kc,    // updated key cache (S,H,DHEAD)
                 const float* __restrict__ vc,    // updated value cache
                 float* __restrict__ attn) {      // (H, DHEAD)
    const int h = blockIdx.x;
    const int t = threadIdx.x;                    // 0..127
    const float scale = rsqrtf((float)DHEAD);

    __shared__ float red[128];
    __shared__ float logits[SEQ];

    for (int j = 0; j < SEQ; ++j) {
        float p = q[h * DHEAD + t] * kc[((size_t)j * HEADS + h) * DHEAD + t];
        red[t] = p;
        __syncthreads();
        for (int s = 64; s > 0; s >>= 1) {
            if (t < s) red[t] += red[t + s];
            __syncthreads();
        }
        if (t == 0) logits[j] = red[0] * scale;
        __syncthreads();
    }

    float m = -INFINITY;
    for (int j = 0; j < SEQ; ++j) m = fmaxf(m, logits[j]);
    float denom = 0.0f;
    float w[SEQ];
    for (int j = 0; j < SEQ; ++j) { w[j] = __expf(logits[j] - m); denom += w[j]; }
    const float inv_denom = 1.0f / denom;

    float acc = 0.0f;
    for (int j = 0; j < SEQ; ++j)
        acc += (w[j] * inv_denom) * vc[((size_t)j * HEADS + h) * DHEAD + t];

    attn[h * DHEAD + t] = acc;
}

// ---------------------------------------------------------------------------
// Launch. d_out layout (tuple flattened): out[4096] | key_cache | value_cache.
// d_ws layout: q[4096] | k[4096] | v[4096] | attn[4096]  (64 KB).
// ---------------------------------------------------------------------------
extern "C" void kernel_launch(void* const* d_in, const int* in_sizes, int n_in,
                              void* d_out, int out_size, void* d_ws, size_t ws_size,
                              hipStream_t stream) {
    const float* x    = (const float*)d_in[0];
    const float* Wq   = (const float*)d_in[1];
    const float* Wk   = (const float*)d_in[2];
    const float* Wv   = (const float*)d_in[3];
    const float* Wo   = (const float*)d_in[4];
    const float* kc   = (const float*)d_in[5];
    const float* vc   = (const float*)d_in[6];
    const int*   idxp = (const int*)d_in[7];

    float* out  = (float*)d_out;                      // 4096
    float* kout = out + DOUT;                         // S*H*DK
    float* vout = kout + (size_t)SLEN * HEADS * DHEAD;

    float* q    = (float*)d_ws;
    float* k    = q + HEADS * DHEAD;
    float* v    = k + HEADS * DHEAD;
    float* attn = v + HEADS * DHEAD;

    // q/k/v projections: 4096 outputs each, 16 outputs per wave-block.
    wmma_matvec_kernel<<<DIN / 16, 32, 0, stream>>>(x, Wq, q, DIN);
    wmma_matvec_kernel<<<DIN / 16, 32, 0, stream>>>(x, Wk, k, DIN);
    wmma_matvec_kernel<<<DIN / 16, 32, 0, stream>>>(x, Wv, v, DIN);

    // RoPE (positions 0..SEQ-1).
    rope_kernel<<<HEADS, 64, 0, stream>>>(q, k);

    // Streamed cache copy (async-LDS path on CDNA5), then row update at index.
    copy_caches_kernel<<<2048, 256, 0, stream>>>(
        (const float4*)kc, (const float4*)vc, (float4*)kout, (float4*)vout);
    update_row_kernel<<<8, 256, 0, stream>>>(
        (const float4*)k, (const float4*)v, idxp, (float4*)kout, (float4*)vout);

    // Masked softmax attention over the valid prefix (uses updated caches).
    attn_kernel<<<HEADS, 128, 0, stream>>>(q, kout, vout, attn);

    // Output projection into d_out[0:4096].
    wmma_matvec_kernel<<<DOUT / 16, 32, 0, stream>>>(attn, Wo, out, DIN);
}
